// MultiRegionRNN_42099269436205
// MI455X (gfx1250) — matrix operand order repaired and checked by reading, not compiled, not run on hardware
//
#include <hip/hip_runtime.h>
#include <hip/hip_bf16.h>
#include <math.h>

typedef __attribute__((ext_vector_type(16))) _Float16 v16h;
typedef __attribute__((ext_vector_type(8)))  _Float16 v8h;
typedef __attribute__((ext_vector_type(8)))  float    v8f;

#define NREG 512
#define BROWS 64
#define TSTEPS 300
#define DIN 10
#define DOUT 2
#define REG_ELEMS (BROWS * NREG)          // 32768
#define ALPHA_F 0.2f
#define ONE_MINUS_ALPHA_F 0.8f

// ---------------------------------------------------------------------------
// Init: zero x state, zero r16 buffer 0 (rates at t=0 are tanh(0)=0), zero
// r_full[0] slices and outs[0] in d_out, build the two pmd bias variants.
// ---------------------------------------------------------------------------
__global__ __launch_bounds__(256) void rnn_init(
    float* __restrict__ x_m1, float* __restrict__ x_pmd, float* __restrict__ x_s1,
    _Float16* __restrict__ r16_buf0,
    float* __restrict__ outs,
    float* __restrict__ r_m1_full, float* __restrict__ r_pmd_full, float* __restrict__ r_s1_full,
    float* __restrict__ bias_pmd_base, float* __restrict__ bias_pmd_full,
    const float* __restrict__ b_m1_pmd, const float* __restrict__ b_s1_pmd,
    const float* __restrict__ b_fb_pmd, const float* __restrict__ b_out_m1,
    const float* __restrict__ W_fb_pmd)
{
    int idx = blockIdx.x * blockDim.x + threadIdx.x;
    if (idx < REG_ELEMS) {
        x_m1[idx] = 0.f; x_pmd[idx] = 0.f; x_s1[idx] = 0.f;
        r16_buf0[idx] = (_Float16)0.f;
        r16_buf0[idx + REG_ELEMS] = (_Float16)0.f;
        r16_buf0[idx + 2 * REG_ELEMS] = (_Float16)0.f;
        r_m1_full[idx] = 0.f; r_pmd_full[idx] = 0.f; r_s1_full[idx] = 0.f;
    }
    if (idx < BROWS * DOUT) outs[idx] = 0.f;
    if (idx < NREG) {
        float base = b_m1_pmd[idx] + b_s1_pmd[idx] + b_fb_pmd[idx];
        bias_pmd_base[idx] = base;
        bias_pmd_full[idx] = base + b_out_m1[0] * W_fb_pmd[idx]
                                  + b_out_m1[1] * W_fb_pmd[NREG + idx];
    }
}

// ---------------------------------------------------------------------------
// Weight prep: f32 row-major [K][N] -> f16 transposed [N][K] (B-operand
// friendly: contiguous K per output column). Also folds the rank-2 output
// feedback W_out_m1 @ W_fb_pmd into W_m1_pmd.
// ---------------------------------------------------------------------------
__global__ __launch_bounds__(256) void prep_weights(
    const float* __restrict__ W_rec_m1, const float* __restrict__ W_rec_pmd,
    const float* __restrict__ W_rec_s1, const float* __restrict__ W_pmd_m1,
    const float* __restrict__ W_m1_pmd, const float* __restrict__ W_s1_pmd,
    const float* __restrict__ W_out_m1, const float* __restrict__ W_fb_pmd,
    _Float16* __restrict__ Wt_rec_m1, _Float16* __restrict__ Wt_rec_pmd,
    _Float16* __restrict__ Wt_rec_s1, _Float16* __restrict__ Wt_pmd_m1,
    _Float16* __restrict__ Wt_m1_pmd_eff, _Float16* __restrict__ Wt_s1_pmd)
{
    int idx = blockIdx.x * blockDim.x + threadIdx.x;
    if (idx >= NREG * NREG) return;
    int n = idx >> 9;          // output column (row of transposed)
    int k = idx & (NREG - 1);  // reduction index
    size_t src = (size_t)k * NREG + n;
    size_t dst = (size_t)n * NREG + k;
    Wt_rec_m1[dst] = (_Float16)W_rec_m1[src];
    Wt_rec_pmd[dst] = (_Float16)W_rec_pmd[src];
    Wt_rec_s1[dst] = (_Float16)W_rec_s1[src];
    Wt_pmd_m1[dst] = (_Float16)W_pmd_m1[src];
    Wt_s1_pmd[dst] = (_Float16)W_s1_pmd[src];
    float fold = W_out_m1[k * DOUT + 0] * W_fb_pmd[n]
               + W_out_m1[k * DOUT + 1] * W_fb_pmd[NREG + n];
    Wt_m1_pmd_eff[dst] = (_Float16)(W_m1_pmd[src] + fold);
}

// ---------------------------------------------------------------------------
// K-loop GEMM accumulate for one 16x16 tile. Pointers trace straight back to
// kernel args so infer-address-spaces promotes every load to global_load.
//   A fragment (16-bit A 16x32): lane holds row M=lane%16;
//     elems 0..15 -> K = kc + 16*(lane/16) + {0..15} (contiguous).
//   B fragment (transposed weights): lane holds col N=n0+lane%16,
//     elems 0..15 -> K = kc + 16*(lane/16) + {0..15} (contiguous).
// ---------------------------------------------------------------------------
__device__ __forceinline__ v8f gemm_acc(
    v8f acc, const _Float16* __restrict__ Abase, const _Float16* __restrict__ Bbase,
    int m0, int n0, int lmod, int lhalf)
{
    const _Float16* Arow = Abase + (size_t)(m0 + lmod) * NREG + lhalf * 16;
    const _Float16* Brow = Bbase + (size_t)(n0 + lmod) * NREG + lhalf * 16;
    #pragma unroll
    for (int kc = 0; kc < NREG; kc += 32) {
        v8h a_lo = *(const v8h*)(Arow + kc);
        v8h a_hi = *(const v8h*)(Arow + kc + 8);
        v16h afrag = __builtin_shufflevector(a_lo, a_hi,
            0, 1, 2, 3, 4, 5, 6, 7, 8, 9, 10, 11, 12, 13, 14, 15);
        v16h bfrag = *(const v16h*)(Brow + kc);
        acc = __builtin_amdgcn_wmma_f32_16x16x32_f16(
            false, afrag, false, bfrag, (short)0, acc, false, false);
    }
    return acc;
}

// ---------------------------------------------------------------------------
// One recurrent timestep. 384 waves, one 16x16 output tile per wave.
//   region 0 (m1):  acc = r_m1@W_rec_m1 + r_pmd@W_pmd_m1
//   region 1 (pmd): acc = r_m1@W_m1_pmd_eff + r_pmd@W_rec_pmd + r_s1@W_s1_pmd
//   region 2 (s1):  acc = r_s1@W_rec_s1 (+ scalar stim@W_in_s1)
//   x <- 0.8 x + 0.2 (acc + bias);  r = tanh(x)
// ---------------------------------------------------------------------------
__global__ __launch_bounds__(256) void rnn_step(
    const _Float16* __restrict__ Wt_rec_m1, const _Float16* __restrict__ Wt_rec_pmd,
    const _Float16* __restrict__ Wt_rec_s1, const _Float16* __restrict__ Wt_pmd_m1,
    const _Float16* __restrict__ Wt_m1_pmd_eff, const _Float16* __restrict__ Wt_s1_pmd,
    const float* __restrict__ biasPmd,       // host-selected: base (t==1) / full (t>=2)
    const float* __restrict__ b_pmd_m1,
    const float* __restrict__ b_in_s1,
    const float* __restrict__ stim_t,        // stim + t*B*DIN
    const float* __restrict__ W_in_s1,       // [DIN][N] f32
    float* __restrict__ x_m1, float* __restrict__ x_pmd, float* __restrict__ x_s1,
    const _Float16* __restrict__ r16_in,     // [3][B][N] rates at t-1
    _Float16* __restrict__ r16_out,          // [3][B][N] rates at t
    float* __restrict__ rout_m1, float* __restrict__ rout_pmd, float* __restrict__ rout_s1)
{
    const int wave = (blockIdx.x * blockDim.x + threadIdx.x) >> 5; // 0..383
    const int lane = threadIdx.x & 31;
    const int region = wave >> 7;        // 0..2
    const int tile = wave & 127;
    const int m0 = (tile >> 5) * 16;     // 0,16,32,48
    const int n0 = (tile & 31) * 16;     // 0..496
    const int lhalf = lane >> 4;         // 0/1
    const int lmod = lane & 15;

    const _Float16* r_m1 = r16_in;
    const _Float16* r_pmd = r16_in + REG_ELEMS;
    const _Float16* r_s1 = r16_in + 2 * REG_ELEMS;

    v8f acc = {};
    if (region == 0) {
        acc = gemm_acc(acc, r_m1, Wt_rec_m1, m0, n0, lmod, lhalf);
        acc = gemm_acc(acc, r_pmd, Wt_pmd_m1, m0, n0, lmod, lhalf);
    } else if (region == 1) {
        acc = gemm_acc(acc, r_m1, Wt_m1_pmd_eff, m0, n0, lmod, lhalf);
        acc = gemm_acc(acc, r_pmd, Wt_rec_pmd, m0, n0, lmod, lhalf);
        acc = gemm_acc(acc, r_s1, Wt_s1_pmd, m0, n0, lmod, lhalf);
    } else {
        acc = gemm_acc(acc, r_s1, Wt_rec_s1, m0, n0, lmod, lhalf);
    }

    // Scalar input projection for S1 tiles: stim[t] @ W_in_s1 (K = 10).
    float sextra[8] = {0.f, 0.f, 0.f, 0.f, 0.f, 0.f, 0.f, 0.f};
    const int N = n0 + lmod;
    if (region == 2) {
        float wcol[DIN];
        #pragma unroll
        for (int k = 0; k < DIN; ++k) wcol[k] = W_in_s1[k * NREG + N];
        #pragma unroll
        for (int i = 0; i < 8; ++i) {
            const int M = m0 + i + lhalf * 8;
            const float* srow = stim_t + M * DIN;
            float sv = 0.f;
            #pragma unroll
            for (int k = 0; k < DIN; ++k) sv += srow[k] * wcol[k];
            sextra[i] = sv;
        }
    }

    float bias;
    float* xbuf;
    _Float16* r16o;
    float* routo;
    if (region == 0)      { bias = b_pmd_m1[N]; xbuf = x_m1;  r16o = r16_out;                 routo = rout_m1; }
    else if (region == 1) { bias = biasPmd[N];  xbuf = x_pmd; r16o = r16_out + REG_ELEMS;     routo = rout_pmd; }
    else                  { bias = b_in_s1[N];  xbuf = x_s1;  r16o = r16_out + 2 * REG_ELEMS; routo = rout_s1; }

    // C/D layout: elem i -> M = m0 + i + 8*lhalf, N = n0 + lane%16
    #pragma unroll
    for (int i = 0; i < 8; ++i) {
        const int M = m0 + i + lhalf * 8;
        const size_t idx = (size_t)M * NREG + N;
        float xn = ONE_MINUS_ALPHA_F * xbuf[idx]
                 + ALPHA_F * (acc[i] + sextra[i] + bias);
        xbuf[idx] = xn;
        float r = tanhf(xn);
        routo[idx] = r;
        r16o[idx] = (_Float16)r;
    }
}

// ---------------------------------------------------------------------------
// Epilogue: outs[t] = r_m1_full[t] @ W_out_m1 + b_out, t = 1..T-1 (batched).
// ---------------------------------------------------------------------------
__global__ __launch_bounds__(256) void rnn_out(
    const float* __restrict__ r_m1_full, const float* __restrict__ W_out_m1,
    const float* __restrict__ b_out_m1, float* __restrict__ outs)
{
    int idx = blockIdx.x * blockDim.x + threadIdx.x;
    if (idx >= (TSTEPS - 1) * BROWS * DOUT) return;
    const int d = idx & 1;
    const int tb = idx >> 1;
    const int b = tb & (BROWS - 1);
    const int t = (tb >> 6) + 1;
    const float* row = r_m1_full + ((size_t)t * BROWS + b) * NREG;
    float s = b_out_m1[d];
    #pragma unroll 8
    for (int k = 0; k < NREG; ++k) s += row[k] * W_out_m1[k * DOUT + d];
    outs[((size_t)t * BROWS + b) * DOUT + d] = s;
}

// ---------------------------------------------------------------------------
extern "C" void kernel_launch(void* const* d_in, const int* in_sizes, int n_in,
                              void* d_out, int out_size, void* d_ws, size_t ws_size,
                              hipStream_t stream) {
    const float* stim      = (const float*)d_in[0];
    const float* W_rec_m1  = (const float*)d_in[1];
    const float* W_rec_pmd = (const float*)d_in[2];
    const float* W_rec_s1  = (const float*)d_in[3];
    const float* W_pmd_m1  = (const float*)d_in[4];
    const float* b_pmd_m1  = (const float*)d_in[5];
    const float* W_m1_pmd  = (const float*)d_in[6];
    const float* b_m1_pmd  = (const float*)d_in[7];
    const float* W_s1_pmd  = (const float*)d_in[8];
    const float* b_s1_pmd  = (const float*)d_in[9];
    const float* W_in_s1   = (const float*)d_in[10];
    const float* b_in_s1   = (const float*)d_in[11];
    const float* W_out_m1  = (const float*)d_in[12];
    const float* b_out_m1  = (const float*)d_in[13];
    const float* W_fb_pmd  = (const float*)d_in[14];
    const float* b_fb_pmd  = (const float*)d_in[15];

    // d_out layout: outs [300,64,2] | r_m1 [300,64,512] | r_pmd | r_s1
    float* outs = (float*)d_out;
    float* r_m1_full = outs + (size_t)TSTEPS * BROWS * DOUT;
    float* r_pmd_full = r_m1_full + (size_t)TSTEPS * REG_ELEMS;
    float* r_s1_full = r_pmd_full + (size_t)TSTEPS * REG_ELEMS;

    // Workspace carve-up (~3.76 MB)
    char* ws = (char*)d_ws;
    const size_t WSZ = (size_t)NREG * NREG * sizeof(_Float16); // 512 KB each
    _Float16* Wt_rec_m1     = (_Float16*)(ws + 0 * WSZ);
    _Float16* Wt_rec_pmd    = (_Float16*)(ws + 1 * WSZ);
    _Float16* Wt_rec_s1     = (_Float16*)(ws + 2 * WSZ);
    _Float16* Wt_pmd_m1     = (_Float16*)(ws + 3 * WSZ);
    _Float16* Wt_m1_pmd_eff = (_Float16*)(ws + 4 * WSZ);
    _Float16* Wt_s1_pmd     = (_Float16*)(ws + 5 * WSZ);
    float* bias_pmd_base = (float*)(ws + 6 * WSZ);
    float* bias_pmd_full = bias_pmd_base + NREG;
    float* x_m1 = (float*)(ws + 6 * WSZ + 4096);
    float* x_pmd = x_m1 + REG_ELEMS;
    float* x_s1 = x_pmd + REG_ELEMS;
    _Float16* r16 = (_Float16*)(x_s1 + REG_ELEMS); // 2 buffers x 3 regions x 32768 f16

    rnn_init<<<128, 256, 0, stream>>>(
        x_m1, x_pmd, x_s1, r16, outs, r_m1_full, r_pmd_full, r_s1_full,
        bias_pmd_base, bias_pmd_full,
        b_m1_pmd, b_s1_pmd, b_fb_pmd, b_out_m1, W_fb_pmd);

    prep_weights<<<1024, 256, 0, stream>>>(
        W_rec_m1, W_rec_pmd, W_rec_s1, W_pmd_m1, W_m1_pmd, W_s1_pmd,
        W_out_m1, W_fb_pmd,
        Wt_rec_m1, Wt_rec_pmd, Wt_rec_s1, Wt_pmd_m1, Wt_m1_pmd_eff, Wt_s1_pmd);

    for (int t = 1; t < TSTEPS; ++t) {
        _Float16* rin = r16 + (size_t)((t + 1) & 1) * (3 * REG_ELEMS);
        _Float16* rout = r16 + (size_t)(t & 1) * (3 * REG_ELEMS);
        const float* biasPmd = (t == 1) ? bias_pmd_base : bias_pmd_full;
        rnn_step<<<48, 256, 0, stream>>>(
            Wt_rec_m1, Wt_rec_pmd, Wt_rec_s1, Wt_pmd_m1, Wt_m1_pmd_eff, Wt_s1_pmd,
            biasPmd, b_pmd_m1, b_in_s1,
            stim + (size_t)t * BROWS * DIN, W_in_s1,
            x_m1, x_pmd, x_s1, rin, rout,
            r_m1_full + (size_t)t * REG_ELEMS,
            r_pmd_full + (size_t)t * REG_ELEMS,
            r_s1_full + (size_t)t * REG_ELEMS);
    }

    rnn_out<<<150, 256, 0, stream>>>(r_m1_full, W_out_m1, b_out_m1, outs);
}
